// GPT_57569741635984
// MI455X (gfx1250) — matrix operand order repaired and checked by reading, not compiled, not run on hardware
//
#include <hip/hip_runtime.h>
#include <hip/hip_bf16.h>
#include <math.h>

// GPT-2 small config (fixed by the reference)
#define Bn   4
#define Tn   1024
#define Cn   768
#define Hn   12
#define Ln   12
#define Vn   50257
#define HDn  64
#define NTOK (Bn * Tn)          // 4096 tokens

typedef __attribute__((ext_vector_type(16))) _Float16 v16h;
typedef __attribute__((ext_vector_type(8)))  _Float16 v8h;
typedef __attribute__((ext_vector_type(8)))  float    v8f;
typedef __attribute__((ext_vector_type(4)))  int      v4i;

__device__ __forceinline__ v8f wmma_f16(v16h a, v16h b, v8f c) {
  // D = A(16x32 f16) * B(32x16 f16) + C(16x16 f32)
  return __builtin_amdgcn_wmma_f32_16x16x32_f16(
      /*neg_a=*/false, a, /*neg_b=*/false, b,
      /*c_mod=*/(short)0, c, /*reuse_a=*/false, /*reuse_b=*/false);
}

__device__ __forceinline__ v16h cat16(v8h lo, v8h hi) {
  return __builtin_shufflevector(lo, hi, 0, 1, 2, 3, 4, 5, 6, 7,
                                 8, 9, 10, 11, 12, 13, 14, 15);
}

// ---------------------------------------------------------------------------
// Async global -> LDS copy (CDNA5 GLOBAL_LOAD_ASYNC_TO_LDS_B128, ASYNCcnt),
// with a synchronous fallback if the toolchain lacks the builtin.
// Builtin signature (from hipcc diagnostic): (v4i AS1*, v4i AS3*, Imm, Imm).
// ---------------------------------------------------------------------------
#if __has_builtin(__builtin_amdgcn_global_load_async_to_lds_b128)
#define HAS_ASYNC_LDS 1
#else
#define HAS_ASYNC_LDS 0
#endif

typedef v4i __attribute__((address_space(1))) as1_v4i;
typedef v4i __attribute__((address_space(3))) as3_v4i;

__device__ __forceinline__ void async_copy_b128(const void* g, void* l) {
#if HAS_ASYNC_LDS
  // AS(3) pointers are 32-bit; low 32 bits of a generic LDS address are the
  // LDS byte offset (aperture lives in the high bits).
  as1_v4i* gp = (as1_v4i*)(uintptr_t)g;
  as3_v4i* lp = (as3_v4i*)(unsigned)(uintptr_t)l;
  __builtin_amdgcn_global_load_async_to_lds_b128(gp, lp, 0, 0);
#else
  *(v8h*)l = *(const v8h*)g;
#endif
}

__device__ __forceinline__ void wait_async_lds() {
#if HAS_ASYNC_LDS
#if __has_builtin(__builtin_amdgcn_s_wait_asynccnt)
  __builtin_amdgcn_s_wait_asynccnt(0);
#else
  asm volatile("s_wait_asynccnt 0x0" ::: "memory");
#endif
#endif
}

// ---------------------------------------------------------------------------
// fp32 -> f16 weight conversion
// ---------------------------------------------------------------------------
__global__ void __launch_bounds__(256) cvt_f16_kernel(
    const float* __restrict__ in, _Float16* __restrict__ out, int n) {
  int i = blockIdx.x * 256 + threadIdx.x;
  if (i < n) out[i] = (_Float16)in[i];
}

// ---------------------------------------------------------------------------
// x = wte[idx] + wpe   (fp32 residual stream)
// ---------------------------------------------------------------------------
__global__ void __launch_bounds__(256) embed_kernel(
    const int* __restrict__ idx, const float* __restrict__ wte,
    const float* __restrict__ wpe, float* __restrict__ x) {
  int row = blockIdx.x;               // token index 0..4095
  int t   = row & (Tn - 1);
  int tok = idx[row];
  const float* we = wte + (size_t)tok * Cn;
  const float* pe = wpe + (size_t)t * Cn;
  float* xr = x + (size_t)row * Cn;
  for (int c = threadIdx.x; c < Cn; c += 256) xr[c] = we[c] + pe[c];
}

// ---------------------------------------------------------------------------
// LayerNorm: fp32 in -> f16 out  (one block per token row)
// ---------------------------------------------------------------------------
__global__ void __launch_bounds__(256) ln_kernel(
    const float* __restrict__ x, const float* __restrict__ g,
    const float* __restrict__ b, _Float16* __restrict__ out) {
  __shared__ float red[256];
  int row = blockIdx.x;
  const float* xr = x + (size_t)row * Cn;
  float s = 0.f, s2 = 0.f;
  for (int c = threadIdx.x; c < Cn; c += 256) { float v = xr[c]; s += v; s2 += v * v; }
  red[threadIdx.x] = s; __syncthreads();
  for (int o = 128; o > 0; o >>= 1) { if (threadIdx.x < o) red[threadIdx.x] += red[threadIdx.x + o]; __syncthreads(); }
  float mean = red[0] / Cn; __syncthreads();
  red[threadIdx.x] = s2; __syncthreads();
  for (int o = 128; o > 0; o >>= 1) { if (threadIdx.x < o) red[threadIdx.x] += red[threadIdx.x + o]; __syncthreads(); }
  float var = red[0] / Cn - mean * mean;
  float inv = rsqrtf(var + 1e-5f);
  _Float16* orow = out + (size_t)row * Cn;
  for (int c = threadIdx.x; c < Cn; c += 256)
    orow[c] = (_Float16)((xr[c] - mean) * inv * g[c] + b[c]);
}

// ---------------------------------------------------------------------------
// Tiled WMMA GEMM with async-LDS double buffering.
//   C[M,N] = A[M,K](f16) * B[K,N](f16) + bias (+ epilogue)
//   BT:  B stored [N,K] row-major (transposed access, tied LM head)
//   EPI: 0 = fp32 out (+bias)            1 = fp32 out (+bias +residual)
//        2 = f16  out (+bias)            3 = f16  out (+bias, GELU-tanh)
// Block: 256 threads = 8 waves; tile 128x128x32; each wave 32(m) x 64(n).
// ---------------------------------------------------------------------------
template <bool BT, int EPI>
__global__ void __launch_bounds__(256) gemm_kernel(
    const _Float16* __restrict__ A, const _Float16* __restrict__ B,
    const float* __restrict__ bias, const float* __restrict__ resid,
    float* __restrict__ outf, _Float16* __restrict__ outh,
    int M, int N, int K) {
  constexpr int BM = 128, BN = 128, BK = 32, BKP = 40, BNP = 136;
  __shared__ _Float16 As[2][BM * BKP];   // 2 x 10 KiB (80 B pitch, 16B-aligned rows)
  __shared__ _Float16 Bs[2][BK * BNP];   // 2 x 8.5 KiB (272 B pitch, 16B-aligned rows)

  const int tid  = threadIdx.x;
  const int lane = tid & 31;
  const int wave = tid >> 5;
  const int wm   = (wave & 3) * 32;   // wave row offset in tile
  const int wn   = (wave >> 2) * 64;  // wave col offset in tile
  const int bm   = blockIdx.y * BM;
  const int bn   = blockIdx.x * BN;
  const int r15  = lane & 15;
  const int hs8  = (lane >> 4) << 3;  // 0 or 8
  const int hs16 = (lane >> 4) << 4;  // 0 or 16
  const bool edgeN = BT || (bn + BN > N);

  // one 128x32 A tile: each thread copies 16 f16 = 2 x b128
  auto stageA = [&](int buf, int kb) {
    int r = tid >> 1, c16 = (tid & 1) * 16;
    const _Float16* g = &A[(size_t)(bm + r) * K + kb + c16];
    _Float16* l = &As[buf][r * BKP + c16];
    async_copy_b128(g, l);
    async_copy_b128(g + 8, l + 8);
  };
  // one 32x128 B tile
  auto stageB = [&](int buf, int kb) {
    if (!edgeN) {
      int k = tid >> 3, n16 = (tid & 7) * 16;
      const _Float16* g = &B[(size_t)(kb + k) * N + bn + n16];
      _Float16* l = &Bs[buf][k * BNP + n16];
      async_copy_b128(g, l);
      async_copy_b128(g + 8, l + 8);
    } else {  // bounds-checked / transposed path (LM head)
      for (int i = tid; i < BK * BN; i += 256) {
        int r = i >> 7, c = i & 127;            // r = k, c = n
        int gn = bn + c;
        _Float16 v = (_Float16)0.f;
        if (gn < N)
          v = BT ? B[(size_t)gn * K + kb + r] : B[(size_t)(kb + r) * N + gn];
        Bs[buf][r * BNP + c] = v;
      }
    }
  };

  v8f acc[2][4];
  for (int i = 0; i < 2; i++)
    for (int j = 0; j < 4; j++)
      for (int e = 0; e < 8; e++) acc[i][j][e] = 0.f;

  int cur = 0;
  stageA(0, 0);
  stageB(0, 0);
  wait_async_lds();
  __syncthreads();

  for (int kb = 0; kb < K; kb += BK) {
    int nxt = cur ^ 1;
    if (kb + BK < K) {          // prefetch next K-tile while computing
      stageA(nxt, kb + BK);
      stageB(nxt, kb + BK);
    }

    const _Float16* Ab = As[cur];
    const _Float16* Bb = Bs[cur];

    // A fragments: halves {0-15} vs {16-31} -> two b128 LDS loads
    v16h af[2];
    for (int mt = 0; mt < 2; mt++) {
      const _Float16* ap = &Ab[(wm + mt * 16 + r15) * BKP + hs16];
      af[mt] = cat16(*(const v8h*)ap, *(const v8h*)(ap + 8));
    }
    // B fragments: lane = column, lane-half = K 0-15 vs 16-31 (strided)
    v16h bf[4];
    for (int nt = 0; nt < 4; nt++) {
      const _Float16* bp = &Bb[hs16 * BNP + wn + nt * 16 + r15];
      for (int e = 0; e < 16; e++) bf[nt][e] = bp[e * BNP];
    }
    for (int mt = 0; mt < 2; mt++)
      for (int nt = 0; nt < 4; nt++)
        acc[mt][nt] = wmma_f16(af[mt], bf[nt], acc[mt][nt]);

    wait_async_lds();
    __syncthreads();
    cur = nxt;
  }

  // epilogue — C layout: VGPR g -> rows g / g+8 per lane-half, col = lane&15
  for (int mt = 0; mt < 2; mt++)
    for (int nt = 0; nt < 4; nt++)
      for (int g = 0; g < 8; g++) {
        int row = bm + wm + mt * 16 + hs8 + g;
        int col = bn + wn + nt * 16 + r15;
        if (col < N) {
          float v = acc[mt][nt][g];
          if (bias) v += bias[col];
          if (EPI == 1) v += resid[(size_t)row * N + col];
          if (EPI == 3) {
            float u = v;
            v = 0.5f * u * (1.0f + tanhf(0.7978845608028654f * (u + 0.044715f * u * u * u)));
          }
          if (EPI <= 1) outf[(size_t)row * N + col] = v;
          else          outh[(size_t)row * N + col] = (_Float16)v;
        }
      }
}

// ---------------------------------------------------------------------------
// Fused causal flash attention.  qkv: [B*T, 3C] f16 (q|k|v per head inside C)
// grid = (T/128, H, B); 8 waves/block, each wave owns a 16-row Q tile.
// S = Q*K^T and O = P*V via WMMA; online softmax via __shfl_xor within
// 16-lane halves (matches C-layout: rows on VGPR idx + half, cols on lanes).
// ---------------------------------------------------------------------------
__global__ void __launch_bounds__(256) attn_kernel(
    const _Float16* __restrict__ qkv, _Float16* __restrict__ out) {
  __shared__ _Float16 Plds[8 * 16 * 32];
  const int lane  = threadIdx.x & 31;
  const int wave  = threadIdx.x >> 5;
  const int h     = blockIdx.y;
  const int b     = blockIdx.z;
  const int qbase = blockIdx.x * 128 + wave * 16;
  const int r15   = lane & 15;
  const int hs8   = (lane >> 4) << 3;
  const int hs16  = (lane >> 4) << 4;
  const float scale = 0.125f;  // 1/sqrt(64)

  const _Float16* base = qkv + (size_t)b * Tn * (3 * Cn);
  const int qo = h * HDn, ko = Cn + h * HDn, vo = 2 * Cn + h * HDn;
  _Float16* P = &Plds[wave * 16 * 32];

  // Q fragments (two 32-wide d-chunks of hd=64), A-layout, vector loads
  v16h qf[2];
  {
    const _Float16* qrow = base + (size_t)(qbase + r15) * (3 * Cn) + qo + hs16;
    for (int dc = 0; dc < 2; dc++)
      qf[dc] = cat16(*(const v8h*)(qrow + dc * 32),
                     *(const v8h*)(qrow + dc * 32 + 8));
  }

  v8f O[4];
  float mrow[8], lrow[8];
  for (int i = 0; i < 4; i++)
    for (int e = 0; e < 8; e++) O[i][e] = 0.f;
  for (int g = 0; g < 8; g++) { mrow[g] = -1e30f; lrow[g] = 0.f; }

  for (int kc = 0; kc <= qbase + 15; kc += 32) {
    // ---- S tile (16q x 32k), accumulated over d in two K=32 WMMA steps ----
    v8f s[2];
    for (int i = 0; i < 2; i++)
      for (int e = 0; e < 8; e++) s[i][e] = 0.f;
    for (int dc = 0; dc < 2; dc++)
      for (int kt = 0; kt < 2; kt++) {
        // B-layout: col = k-column (lane), rows = d; 16 contiguous halves
        const _Float16* krow =
            base + (size_t)(kc + kt * 16 + r15) * (3 * Cn) + ko + dc * 32 + hs16;
        v16h kf = cat16(*(const v8h*)krow, *(const v8h*)(krow + 8));
        s[kt] = wmma_f16(qf[dc], kf, s[kt]);
      }

    // ---- scale + causal mask ----
    for (int kt = 0; kt < 2; kt++)
      for (int g = 0; g < 8; g++) {
        float v = s[kt][g] * scale;
        int q = qbase + hs8 + g;
        int k = kc + kt * 16 + r15;
        s[kt][g] = (k > q) ? -1e30f : v;
      }

    // ---- online softmax (row stats reduced across 16-lane halves) ----
    float alpha[8];
    for (int g = 0; g < 8; g++) {
      float mx = fmaxf(s[0][g], s[1][g]);
      for (int off = 1; off < 16; off <<= 1) mx = fmaxf(mx, __shfl_xor(mx, off, 32));
      float mn = fmaxf(mrow[g], mx);
      alpha[g] = __expf(mrow[g] - mn);
      mrow[g]  = mn;
      float p0 = __expf(s[0][g] - mn);
      float p1 = __expf(s[1][g] - mn);
      s[0][g] = p0; s[1][g] = p1;
      float rs = p0 + p1;
      for (int off = 1; off < 16; off <<= 1) rs += __shfl_xor(rs, off, 32);
      lrow[g] = lrow[g] * alpha[g] + rs;
    }
    for (int dt = 0; dt < 4; dt++)
      for (int g = 0; g < 8; g++) O[dt][g] *= alpha[g];

    // ---- re-layout P: C-format regs -> LDS -> A-format frag ----
    for (int g = 0; g < 8; g++) {
      int m = hs8 + g;
      P[m * 32 + r15]      = (_Float16)s[0][g];
      P[m * 32 + 16 + r15] = (_Float16)s[1][g];
    }
    __asm__ volatile("s_wait_dscnt 0x0" ::: "memory");
    const _Float16* pr = &P[r15 * 32 + hs16];
    v16h pf = cat16(*(const v8h*)pr, *(const v8h*)(pr + 8));

    // ---- O += P * V ----
    for (int dt = 0; dt < 4; dt++) {
      v16h vf;  // B-layout: col = d (lane), rows = k (strided by 3C)
      const _Float16* vp =
          base + (size_t)(kc + hs16) * (3 * Cn) + vo + dt * 16 + r15;
      for (int e = 0; e < 16; e++) vf[e] = vp[(size_t)e * (3 * Cn)];
      O[dt] = wmma_f16(pf, vf, O[dt]);
    }
  }

  // ---- finalize: O /= l, store f16 [B*T, C] at this head's slice ----
  for (int dt = 0; dt < 4; dt++)
    for (int g = 0; g < 8; g++) {
      int qrow = qbase + hs8 + g;
      int col  = h * HDn + dt * 16 + r15;
      out[(size_t)(b * Tn + qrow) * Cn + col] = (_Float16)(O[dt][g] / lrow[g]);
    }
}

// ---------------------------------------------------------------------------
// Host driver
// ---------------------------------------------------------------------------
extern "C" void kernel_launch(void* const* d_in, const int* in_sizes, int n_in,
                              void* d_out, int out_size, void* d_ws, size_t ws_size,
                              hipStream_t stream) {
  (void)in_sizes; (void)n_in; (void)out_size; (void)ws_size;
  const int*   idx    = (const int*)  d_in[0];
  const float* wte    = (const float*)d_in[1];
  const float* wpe    = (const float*)d_in[2];
  const float* w_attn = (const float*)d_in[3];
  const float* b_attn = (const float*)d_in[4];
  const float* w_proj = (const float*)d_in[5];
  const float* b_proj = (const float*)d_in[6];
  const float* ln1_g  = (const float*)d_in[7];
  const float* ln1_b  = (const float*)d_in[8];
  const float* ln2_g  = (const float*)d_in[9];
  const float* ln2_b  = (const float*)d_in[10];
  const float* w_fc   = (const float*)d_in[11];
  const float* b_fc   = (const float*)d_in[12];
  const float* w_fc2  = (const float*)d_in[13];
  const float* b_fc2  = (const float*)d_in[14];
  const float* lnf_g  = (const float*)d_in[15];
  const float* lnf_b  = (const float*)d_in[16];
  const float* lm_b   = (const float*)d_in[17];
  float* logits = (float*)d_out;

  // workspace carve-up
  char* ws = (char*)d_ws;
  auto carve = [&](size_t bytes) {
    char* p = ws;
    ws += (bytes + 255) & ~(size_t)255;
    return p;
  };
  const size_t nWte  = (size_t)Vn * Cn;
  const size_t nWa   = (size_t)Ln * Cn * 3 * Cn;
  const size_t nWp   = (size_t)Ln * Cn * Cn;
  const size_t nWf   = (size_t)Ln * Cn * 4 * Cn;
  _Float16* wte_h  = (_Float16*)carve(nWte * 2);
  _Float16* wa_h   = (_Float16*)carve(nWa * 2);
  _Float16* wp_h   = (_Float16*)carve(nWp * 2);
  _Float16* wf_h   = (_Float16*)carve(nWf * 2);
  _Float16* wf2_h  = (_Float16*)carve(nWf * 2);
  float*    x      = (float*)   carve((size_t)NTOK * Cn * 4);
  _Float16* hbuf   = (_Float16*)carve((size_t)NTOK * Cn * 2);
  _Float16* qkvb   = (_Float16*)carve((size_t)NTOK * 3 * Cn * 2);
  _Float16* attnb  = (_Float16*)carve((size_t)NTOK * Cn * 2);
  _Float16* fcb    = (_Float16*)carve((size_t)NTOK * 4 * Cn * 2);
  _Float16* xf     = (_Float16*)carve((size_t)NTOK * Cn * 2);

  // 1) weights fp32 -> f16
  auto cvt = [&](const float* src, _Float16* dst, size_t n) {
    cvt_f16_kernel<<<dim3((unsigned)((n + 255) / 256)), 256, 0, stream>>>(src, dst, (int)n);
  };
  cvt(wte,    wte_h, nWte);
  cvt(w_attn, wa_h,  nWa);
  cvt(w_proj, wp_h,  nWp);
  cvt(w_fc,   wf_h,  nWf);
  cvt(w_fc2,  wf2_h, nWf);

  // 2) embedding
  embed_kernel<<<NTOK, 256, 0, stream>>>(idx, wte, wpe, x);

  const dim3 gQKV((3 * Cn) / 128, NTOK / 128);
  const dim3 gPRJ(Cn / 128,       NTOK / 128);
  const dim3 gFC((4 * Cn) / 128,  NTOK / 128);
  const dim3 gATT(Tn / 128, Hn, Bn);
  const dim3 gLM((Vn + 127) / 128, NTOK / 128);

  for (int l = 0; l < Ln; l++) {
    const _Float16* wa  = wa_h  + (size_t)l * Cn * 3 * Cn;
    const _Float16* wp  = wp_h  + (size_t)l * Cn * Cn;
    const _Float16* wf  = wf_h  + (size_t)l * Cn * 4 * Cn;
    const _Float16* wf2 = wf2_h + (size_t)l * 4 * Cn * Cn;

    // h = LN1(x)
    ln_kernel<<<NTOK, 256, 0, stream>>>(x, ln1_g + (size_t)l * Cn, ln1_b + (size_t)l * Cn, hbuf);
    // qkv = h @ w_attn + b_attn   (f16 out)
    gemm_kernel<false, 2><<<gQKV, 256, 0, stream>>>(
        hbuf, wa, b_attn + (size_t)l * 3 * Cn, nullptr, nullptr, qkvb,
        NTOK, 3 * Cn, Cn);
    // attention
    attn_kernel<<<gATT, 256, 0, stream>>>(qkvb, attnb);
    // x = x + attn @ w_proj + b_proj   (fp32, in place)
    gemm_kernel<false, 1><<<gPRJ, 256, 0, stream>>>(
        attnb, wp, b_proj + (size_t)l * Cn, x, x, nullptr,
        NTOK, Cn, Cn);
    // h = LN2(x)
    ln_kernel<<<NTOK, 256, 0, stream>>>(x, ln2_g + (size_t)l * Cn, ln2_b + (size_t)l * Cn, hbuf);
    // fcact = gelu(h @ w_fc + b_fc)   (f16 out)
    gemm_kernel<false, 3><<<gFC, 256, 0, stream>>>(
        hbuf, wf, b_fc + (size_t)l * 4 * Cn, nullptr, nullptr, fcb,
        NTOK, 4 * Cn, Cn);
    // x = x + fcact @ w_fc2 + b_fc2   (fp32, in place)
    gemm_kernel<false, 1><<<gPRJ, 256, 0, stream>>>(
        fcb, wf2, b_fc2 + (size_t)l * Cn, x, x, nullptr,
        NTOK, Cn, 4 * Cn);
  }

  // final LN, tied LM head: logits = LN_f(x) @ wte^T + lm_b
  ln_kernel<<<NTOK, 256, 0, stream>>>(x, lnf_g, lnf_b, xf);
  gemm_kernel<true, 0><<<gLM, 256, 0, stream>>>(
      xf, wte_h, lm_b, nullptr, logits, nullptr,
      NTOK, Vn, Cn);
}
